// KDAPolicyNetwork_28389733827235
// MI455X (gfx1250) — matrix operand order repaired and checked by reading, not compiled, not run on hardware
//
#include <hip/hip_runtime.h>
#include <math.h>

// ---------------------------------------------------------------------------
// Static problem dims (from the reference)
// ---------------------------------------------------------------------------
#define BB    16
#define TT    4096
#define BT    65536          // BB*TT tokens
#define DD    32             // model dim
#define NN    4              // streams
#define NDD   128            // N*D
#define EPS_RMS 1.1920929e-07f
#define TWO_PI 6.283185307179586f
#define WPB   4              // waves (16-row tiles) per GEMM block
#define LPITCH 136           // LDS row pitch in bf16 elems (16B aligned, bank-staggered)

typedef __bf16 bf16_t;
typedef bf16_t v16bf __attribute__((ext_vector_type(16)));
typedef bf16_t v8bf  __attribute__((ext_vector_type(8)));
typedef float  v8f   __attribute__((ext_vector_type(8)));

#if defined(__HIP_DEVICE_COMPILE__) && __has_builtin(__builtin_amdgcn_tensor_load_to_lds) && __has_builtin(__builtin_amdgcn_s_wait_tensorcnt)
#define HAVE_TDM 1
#else
#define HAVE_TDM 0
#endif

#if HAVE_TDM
typedef unsigned int u32x4 __attribute__((ext_vector_type(4)));
typedef int          i32x4 __attribute__((ext_vector_type(4)));
typedef int          i32x8 __attribute__((ext_vector_type(8)));

// 1-D contiguous DMA: nElems f32 from gptr -> LDS byte offset ldsOff.
// D# group0: count=1, lds_addr, 57b global addr, type=2 ("image").
// D# group1: data_size=4B, tensor_dim0=tile_dim0=nElems, dims1..=1.
__device__ __forceinline__ void tdm_load_1d(const float* gptr, unsigned ldsOff, int nElems)
{
    unsigned long long ga = (unsigned long long)(uintptr_t)gptr;
    u32x4 g0 = {};
    g0[0] = 1u;                                               // count=1
    g0[1] = ldsOff;                                           // lds_addr (bytes)
    g0[2] = (unsigned)(ga & 0xffffffffu);                     // global_addr[31:0]
    g0[3] = (unsigned)((ga >> 32) & 0x1ffffffu) | (2u << 30); // addr[56:32] | type=2
    i32x8 g1 = {};
    g1[0] = (2 << 16);                                        // data_size = 4 bytes
    g1[1] = (nElems & 0xffff) << 16;                          // tensor_dim0[15:0]
    g1[2] = ((nElems >> 16) & 0xffff) | (1 << 16);            // dim0[31:16] | tensor_dim1=1
    g1[3] = ((nElems & 0xffff) << 16);                        // tile_dim0 = nElems
    g1[4] = 1;                                                // tile_dim1=1, tile_dim2=0
    g1[5] = nElems;                                           // tensor_dim0_stride lo
    g1[6] = 0;
    g1[7] = 0;
    i32x4 z = {};
#if __clang_major__ >= 23
    i32x8 z8 = {};
    __builtin_amdgcn_tensor_load_to_lds(g0, g1, z, z, z8, 0);
#else
    __builtin_amdgcn_tensor_load_to_lds(g0, g1, z, z, 0);
#endif
}
#endif

__device__ __forceinline__ float sigmoidf_(float x) { return 1.0f / (1.0f + expf(-x)); }

// ---------------------------------------------------------------------------
// Generic GEMM:  C[M,N] = act( A[M,K] @ W[K,N] + bias ), act: 0 none,1 silu,2 sigmoid
// Block = 4 waves = 64 rows x 16 cols. A tile and transposed-W tile staged in
// LDS as zero-padded bf16; hot loop = ds_load_b128 fragments + wmma only.
// M must be a multiple of 64.
// ---------------------------------------------------------------------------
__global__ __launch_bounds__(32 * WPB) void gemm_wmma_bf16(
    const float* __restrict__ A, const float* __restrict__ W,
    const float* __restrict__ bias, float* __restrict__ C,
    int M, int K, int N, int act)
{
    __shared__ bf16_t lA[64 * LPITCH];   // [row][k]  17.4 KB
    __shared__ bf16_t lB[16 * LPITCH];   // [col][k]  (W transposed)  4.4 KB

    const int tid  = threadIdx.x;
    const int wave = tid >> 5;
    const int lane = tid & 31;
    const bool hi  = lane >= 16;
    const int l15  = lane & 15;
    const int row0 = blockIdx.x * 64;            // block's first A/C row
    const int col0 = blockIdx.y * 16;            // block's first C col

    // K padded to power-of-two block count (K here is 9..128)
    const int kshift = (K <= 32) ? 5 : (K <= 64 ? 6 : 7);
    const int KPAD   = 1 << kshift;
    const int kmask  = KPAD - 1;

    // ---- stage A (64 x KPAD) -> bf16, zero-padded, clamped-index loads ----
    for (int idx = tid; idx < (64 << kshift); idx += 32 * WPB) {
        int r = idx >> kshift, k = idx & kmask;
        int ks = (k < K) ? k : (K - 1);
        float v = A[(size_t)(row0 + r) * K + ks];
        lA[r * LPITCH + k] = (bf16_t)((k < K) ? v : 0.0f);
    }
    // ---- stage W^T (16 x KPAD) ----
    for (int idx = tid; idx < (16 << kshift); idx += 32 * WPB) {
        int c = idx >> kshift, k = idx & kmask;
        int cc = col0 + c;
        int ks = (k < K) ? k : (K - 1);
        int cs = (cc < N) ? cc : (N - 1);
        float v = W[(size_t)ks * N + cs];
        lB[c * LPITCH + k] = (bf16_t)((k < K && cc < N) ? v : 0.0f);
    }
    __syncthreads();

    const bf16_t* lAr = &lA[(wave * 16 + l15) * LPITCH];
    const bf16_t* lBr = &lB[l15 * LPITCH];
    const int baseA = hi ? 8 : 0;     // A frag: K {base..base+7, base+16..base+23}
    const int baseB = hi ? 8 : 0;     // B frag: K {base..base+7, base+16..base+23}

    v8f acc = {0.f, 0.f, 0.f, 0.f, 0.f, 0.f, 0.f, 0.f};
    const int kblocks = KPAD >> 5;
    for (int kb = 0; kb < kblocks; ++kb) {
        const int kk = kb << 5;
        v8bf a0 = *(const v8bf*)(lAr + kk + baseA);
        v8bf a1 = *(const v8bf*)(lAr + kk + baseA + 16);
        v8bf b0 = *(const v8bf*)(lBr + kk + baseB);
        v8bf b1 = *(const v8bf*)(lBr + kk + baseB + 16);
        v16bf av = __builtin_shufflevector(a0, a1, 0,1,2,3,4,5,6,7,8,9,10,11,12,13,14,15);
        v16bf bv = __builtin_shufflevector(b0, b1, 0,1,2,3,4,5,6,7,8,9,10,11,12,13,14,15);
        acc = __builtin_amdgcn_wmma_f32_16x16x32_bf16(
            false, av, false, bv, (short)0, acc, false, false);
    }

    const int col = col0 + l15;
    if (col < N) {
        const float bb = bias ? bias[col] : 0.0f;
        const int rbase = row0 + wave * 16 + (hi ? 8 : 0);
#pragma unroll
        for (int vv = 0; vv < 8; ++vv) {
            float y = acc[vv] + bb;
            if (act == 1)      y = y / (1.0f + expf(-y));     // silu
            else if (act == 2) y = 1.0f / (1.0f + expf(-y));  // sigmoid
            C[(size_t)(rbase + vv) * N + col] = y;
        }
    }
}

// ---------------------------------------------------------------------------
// rmsnorm over rows of width Dd
// ---------------------------------------------------------------------------
__global__ void rmsnorm_kernel(const float* __restrict__ in,
                               const float* __restrict__ wv,
                               float* __restrict__ outp,
                               int rows, int Dd, float eps)
{
    int r = blockIdx.x * blockDim.x + threadIdx.x;
    if (r >= rows) return;
    const float* xr = in + (size_t)r * Dd;
    float ss = 0.f;
    for (int i = 0; i < Dd; ++i) { float v = xr[i]; ss += v * v; }
    float inv = rsqrtf(ss / (float)Dd + eps);
    float* orow = outp + (size_t)r * Dd;
    for (int i = 0; i < Dd; ++i) orow[i] = xr[i] * inv * wv[i];
}

// ---------------------------------------------------------------------------
// mHC: per token -> h (H_pre mix), res (H_res mix), H_post
// ---------------------------------------------------------------------------
__global__ void mhc_kernel(const float* __restrict__ snorm,
                           const float* __restrict__ phi_pre,
                           const float* __restrict__ phi_post,
                           const float* __restrict__ phi_res,
                           const float* __restrict__ b_pre,
                           const float* __restrict__ b_post,
                           const float* __restrict__ b_res,
                           const float* __restrict__ a_pre_p,
                           const float* __restrict__ a_post_p,
                           const float* __restrict__ a_res_p,
                           float* __restrict__ h,
                           float* __restrict__ res,
                           float* __restrict__ Hpost_out,
                           int BTtot)
{
    int t = blockIdx.x * blockDim.x + threadIdx.x;
    if (t >= BTtot) return;
    const float* xr = snorm + (size_t)t * NDD;

    float ss = 0.f;
    for (int i = 0; i < NDD; ++i) { float v = xr[i]; ss += v * v; }
    const float inv = rsqrtf(ss / (float)NDD + 1e-8f);

    float accp[4] = {0, 0, 0, 0}, accq[4] = {0, 0, 0, 0};
    float accr[16];
#pragma unroll
    for (int j = 0; j < 16; ++j) accr[j] = 0.f;

    for (int i = 0; i < NDD; ++i) {
        float xv = xr[i] * inv;
#pragma unroll
        for (int j = 0; j < 4; ++j) {
            accp[j] += xv * phi_pre[i * 4 + j];
            accq[j] += xv * phi_post[i * 4 + j];
        }
#pragma unroll
        for (int j = 0; j < 16; ++j) accr[j] += xv * phi_res[i * 16 + j];
    }

    const float a_pre = a_pre_p[0], a_post = a_post_p[0], a_res = a_res_p[0];
    float Hpre[4], Hpost[4];
#pragma unroll
    for (int j = 0; j < 4; ++j) {
        Hpre[j]  = sigmoidf_(a_pre * accp[j] + b_pre[j]);
        Hpost[j] = 2.0f * sigmoidf_(a_post * accq[j] + b_post[j]);
    }

    float Mm[16];
#pragma unroll
    for (int j = 0; j < 16; ++j) Mm[j] = expf(tanhf(a_res * accr[j] + b_res[j]));
#pragma unroll
    for (int it = 0; it < 6; ++it) {
#pragma unroll
        for (int i = 0; i < 4; ++i) {
            float s = Mm[i*4] + Mm[i*4+1] + Mm[i*4+2] + Mm[i*4+3];
            float is = 1.0f / s;
#pragma unroll
            for (int j = 0; j < 4; ++j) Mm[i*4+j] *= is;
        }
#pragma unroll
        for (int j = 0; j < 4; ++j) {
            float s = Mm[j] + Mm[4+j] + Mm[8+j] + Mm[12+j];
            float is = 1.0f / s;
#pragma unroll
            for (int i = 0; i < 4; ++i) Mm[i*4+j] *= is;
        }
    }

    for (int dd = 0; dd < DD; ++dd) {
        float sj0 = xr[dd], sj1 = xr[32+dd], sj2 = xr[64+dd], sj3 = xr[96+dd];
        h[(size_t)t * DD + dd] =
            Hpre[0]*sj0 + Hpre[1]*sj1 + Hpre[2]*sj2 + Hpre[3]*sj3;
#pragma unroll
        for (int n = 0; n < 4; ++n) {
            res[(size_t)t * NDD + n*32 + dd] =
                Mm[n*4]*sj0 + Mm[n*4+1]*sj1 + Mm[n*4+2]*sj2 + Mm[n*4+3]*sj3;
        }
    }
#pragma unroll
    for (int n = 0; n < 4; ++n) Hpost_out[(size_t)t * 4 + n] = Hpost[n];
}

// ---------------------------------------------------------------------------
// PoPE
// ---------------------------------------------------------------------------
__global__ void pope_kernel(const float* __restrict__ lin,
                            const float* __restrict__ delta,
                            float* __restrict__ outp, int BTtot, int Tlen)
{
    int t = blockIdx.x * blockDim.x + threadIdx.x;
    if (t >= BTtot) return;
    float x[16]; float ss = 0.f;
#pragma unroll
    for (int j = 0; j < 16; ++j) { x[j] = lin[(size_t)t * 16 + j]; ss += x[j]*x[j]; }
    float nr = fmaxf(sqrtf(ss), 1e-12f);
    float pos = (float)(t % Tlen);
#pragma unroll
    for (int j = 0; j < 16; ++j) {
        float xv = x[j] / nr;
        float mu = (xv > 20.f) ? xv : log1pf(expf(xv));
        float fr = powf(10000.0f, (float)j * (1.0f / 16.0f));
        float phi = pos * fr;
        if (delta) phi -= TWO_PI * sigmoidf_(delta[j]);
        outp[(size_t)t * 32 + j]      = mu * cosf(phi);
        outp[(size_t)t * 32 + 16 + j] = mu * sinf(phi);
    }
}

// ---------------------------------------------------------------------------
// KDA delta-rule scan.  grid = B, block = 512 (32 lanes = DKP, 16 waves = DV).
// q/k/alpha/v chunks of 64 timesteps are DMA'd into double-buffered LDS by the
// Tensor Data Mover (wave 0 issues; s_wait_tensorcnt + barrier synchronize),
// overlapping the DMA of chunk c+1 with compute on chunk c.
// ---------------------------------------------------------------------------
#define SCH 64
__global__ __launch_bounds__(512) void scan_kernel(
    const float* __restrict__ q, const float* __restrict__ k,
    const float* __restrict__ v, const float* __restrict__ a,
    float* __restrict__ outp, int Tlen)
{
    __shared__ float sQ[2][SCH * 32];
    __shared__ float sK[2][SCH * 32];
    __shared__ float sA[2][SCH * 32];
    __shared__ float sV[2][SCH * 16];

    const int b    = blockIdx.x;
    const int tid  = threadIdx.x;
    const int lane = tid & 31;   // d (DKP)
    const int e    = tid >> 5;   // output dim (DV)
    const float* qb = q + (size_t)b * Tlen * 32;
    const float* kb = k + (size_t)b * Tlen * 32;
    const float* ab = a + (size_t)b * Tlen * 32;
    const float* vb = v + (size_t)b * Tlen * 16;
    const size_t ob16 = (size_t)b * Tlen * 16;
    const int nch = Tlen / SCH;

#if HAVE_TDM
    if (tid < 32) {
        tdm_load_1d(qb, (unsigned)(uintptr_t)&sQ[0][0], SCH * 32);
        tdm_load_1d(kb, (unsigned)(uintptr_t)&sK[0][0], SCH * 32);
        tdm_load_1d(ab, (unsigned)(uintptr_t)&sA[0][0], SCH * 32);
        tdm_load_1d(vb, (unsigned)(uintptr_t)&sV[0][0], SCH * 16);
    }
#endif

    float S = 0.f;
    for (int c = 0; c < nch; ++c) {
        const int cur = c & 1;
#if HAVE_TDM
        if (tid < 32) {
            if (c + 1 < nch) {
                const int nb = cur ^ 1;
                const size_t g32 = (size_t)(c + 1) * SCH * 32;
                const size_t g16 = (size_t)(c + 1) * SCH * 16;
                tdm_load_1d(qb + g32, (unsigned)(uintptr_t)&sQ[nb][0], SCH * 32);
                tdm_load_1d(kb + g32, (unsigned)(uintptr_t)&sK[nb][0], SCH * 32);
                tdm_load_1d(ab + g32, (unsigned)(uintptr_t)&sA[nb][0], SCH * 32);
                tdm_load_1d(vb + g16, (unsigned)(uintptr_t)&sV[nb][0], SCH * 16);
                __builtin_amdgcn_s_wait_tensorcnt(4);   // chunk c complete
            } else {
                __builtin_amdgcn_s_wait_tensorcnt(0);
            }
        }
        __syncthreads();
#else
        // fallback: synchronous cooperative load of this chunk
        __syncthreads();
        for (int i = tid; i < SCH * 32; i += 512) {
            sQ[cur][i] = qb[(size_t)c * SCH * 32 + i];
            sK[cur][i] = kb[(size_t)c * SCH * 32 + i];
            sA[cur][i] = ab[(size_t)c * SCH * 32 + i];
        }
        for (int i = tid; i < SCH * 16; i += 512)
            sV[cur][i] = vb[(size_t)c * SCH * 16 + i];
        __syncthreads();
#endif
        for (int tt = 0; tt < SCH; ++tt) {
            float at = sA[cur][tt * 32 + lane];
            float kt = sK[cur][tt * 32 + lane];
            float qt = sQ[cur][tt * 32 + lane];
            float vt = sV[cur][tt * 16 + e];
            float aS = at * S;
            float r = kt * aS;
#pragma unroll
            for (int m = 16; m > 0; m >>= 1) r += __shfl_xor(r, m, 32);
            S = aS + kt * (vt - r);
            float o = qt * S;
#pragma unroll
            for (int m = 16; m > 0; m >>= 1) o += __shfl_xor(o, m, 32);
            if (lane == 0) outp[ob16 + (size_t)(c * SCH + tt) * 16 + e] = o;
        }
        __syncthreads();   // buffer may be overwritten next iteration
    }
}

// out = rmsnorm(scan_out, post_w) * G2
__global__ void ob_kernel(const float* __restrict__ scano,
                          const float* __restrict__ postw,
                          const float* __restrict__ G2,
                          float* __restrict__ ob, int BTtot)
{
    int t = blockIdx.x * blockDim.x + threadIdx.x;
    if (t >= BTtot) return;
    float x[16]; float ss = 0.f;
#pragma unroll
    for (int e = 0; e < 16; ++e) { x[e] = scano[(size_t)t*16 + e]; ss += x[e]*x[e]; }
    float inv = rsqrtf(ss / 16.0f + EPS_RMS);
#pragma unroll
    for (int e = 0; e < 16; ++e)
        ob[(size_t)t*16 + e] = x[e] * inv * postw[e] * G2[(size_t)t*16 + e];
}

__global__ void mix_kernel(const float* __restrict__ res,
                           const float* __restrict__ Hpost,
                           const float* __restrict__ o32,
                           float* __restrict__ strm, int BTtot)
{
    int i = blockIdx.x * blockDim.x + threadIdx.x;
    if (i >= BTtot * NDD) return;
    int t = i >> 7, r = i & 127, n = r >> 5, d = r & 31;
    strm[i] = res[i] + Hpost[(size_t)t*4 + n] * o32[(size_t)t*32 + d];
}

__global__ void init_stream_kernel(const float* __restrict__ v0,
                                   float* __restrict__ strm, int BTtot)
{
    int i = blockIdx.x * blockDim.x + threadIdx.x;
    if (i >= BTtot * NDD) return;
    int t = i >> 7, r = i & 127;
    strm[i] = (r < 32) ? v0[(size_t)t*32 + r] : 0.0f;
}

__global__ void attnres_kernel(const float* __restrict__ V, int L1,
                               const float* __restrict__ wl,
                               const float* __restrict__ nw,
                               float* __restrict__ dst, int mode, int BTtot)
{
    int t = blockIdx.x * blockDim.x + threadIdx.x;
    if (t >= BTtot) return;
    float lg[9];
    float mx = -1e30f;
    for (int a = 0; a < L1; ++a) {
        const float* Vr = V + ((size_t)a * BTtot + t) * 32;
        float ss = 0.f;
        for (int d = 0; d < 32; ++d) ss += Vr[d] * Vr[d];
        float inv = rsqrtf(ss / 32.0f + EPS_RMS);
        float s = 0.f;
        for (int d = 0; d < 32; ++d) s += wl[d] * Vr[d] * inv * nw[d];
        lg[a] = s; mx = fmaxf(mx, s);
    }
    float se = 0.f;
    for (int a = 0; a < L1; ++a) { lg[a] = expf(lg[a] - mx); se += lg[a]; }
    float ise = 1.0f / se;
    for (int d = 0; d < 32; ++d) {
        float ov = 0.f;
        for (int a = 0; a < L1; ++a) ov += lg[a] * V[((size_t)a * BTtot + t) * 32 + d];
        ov *= ise;
        if (mode == 0) {
#pragma unroll
            for (int n = 0; n < 4; ++n) dst[(size_t)t * NDD + n*32 + d] += ov;
        } else {
            dst[(size_t)t * 32 + d] = ov;
        }
    }
}

__global__ void mul2_kernel(const float* a, const float* b, float* c, int n)
{ int i = blockIdx.x * blockDim.x + threadIdx.x; if (i < n) c[i] = a[i] * b[i]; }

__global__ void mul3_kernel(const float* a, const float* b, const float* c, float* d, int n)
{ int i = blockIdx.x * blockDim.x + threadIdx.x; if (i < n) d[i] = a[i] * b[i] * c[i]; }

// ---------------------------------------------------------------------------
// Host orchestration
// ---------------------------------------------------------------------------
static void gemm(const float* A, const float* W, const float* bias, float* C,
                 int M, int K, int N, int act, hipStream_t s)
{
    dim3 g((unsigned)(M / 64), (unsigned)((N + 15) / 16), 1);
    gemm_wmma_bf16<<<g, 32 * WPB, 0, s>>>(A, W, bias, C, M, K, N, act);
}

static inline int nblk(int n) { return (n + 255) / 256; }

extern "C" void kernel_launch(void* const* d_in, const int* in_sizes, int n_in,
                              void* d_out, int out_size, void* d_ws, size_t ws_size,
                              hipStream_t stream)
{
    (void)in_sizes; (void)n_in; (void)out_size; (void)ws_size;
    // JAX pytree (sorted-key) leaf order: params (sorted) then x.
#define P(i) ((const float*)d_in[(i)])
    const int KDA0 = 7, MHC0 = 95, PREH0 = 239, SNW0 = 247, SWI0 = 263, WW0 = 311, XIDX = 319;

    float* wsf = (float*)d_ws;
    size_t off = 0;
    auto nxt = [&](size_t n) { float* p = wsf + off; off += n; return p; };
    const size_t BTL = BT;
    float* strmbuf = nxt(BTL * 128);
    float* snorm   = nxt(BTL * 128);
    float* resbuf  = nxt(BTL * 128);
    float* hbuf    = nxt(BTL * 32);
    float* Hpost   = nxt(BTL * 4);
    float* qlin    = nxt(BTL * 16);
    float* klin    = nxt(BTL * 16);
    float* qb      = nxt(BTL * 32);
    float* kb      = nxt(BTL * 32);
    float* vbuf    = nxt(BTL * 16);
    float* AG      = nxt(BTL * 25);
    float* AU      = nxt(BTL * 25);
    float* alphab  = nxt(BTL * 32);
    float* DG      = nxt(BTL * 9);
    float* G2      = nxt(BTL * 16);
    float* scano   = nxt(BTL * 16);
    float* obuf    = nxt(BTL * 16);
    float* o32     = nxt(BTL * 32);
    float* hn      = nxt(BTL * 32);
    float* gd      = nxt(BTL * 32);
    float* gbuf    = nxt(BTL * 51);
    float* t1      = nxt(BTL * 51);
    float* t2      = nxt(BTL * 51);
    float* hout    = nxt(BTL * 32);
    float* Vbuf    = nxt(BTL * 32 * 9);
    float* accb    = nxt(BTL * 128 * 9);

    // ---- input embed: v0 = rmsnorm(silu(x@W_in + b_in)) ----
    gemm(P(XIDX), P(0), P(2), o32, BT, 14, 32, 1, stream);
    rmsnorm_kernel<<<nblk(BT), 256, 0, stream>>>(o32, P(6), hn, BT, 32, EPS_RMS);
    init_stream_kernel<<<nblk(BT * 128), 256, 0, stream>>>(hn, strmbuf, BT);
    hipMemcpyAsync(accb, strmbuf, (size_t)BT * 128 * sizeof(float),
                   hipMemcpyDeviceToDevice, stream);

    for (int l = 0; l < 8; ++l) {
        // ================= KDA sublayer =================
        rmsnorm_kernel<<<nblk(BT * 4), 256, 0, stream>>>(
            strmbuf, P(SNW0 + 2 * l), snorm, BT * 4, 32, EPS_RMS);
        {
            int m = MHC0 + 9 * (2 * l);
            mhc_kernel<<<nblk(BT), 256, 0, stream>>>(
                snorm, P(m + 7), P(m + 6), P(m + 8), P(m + 4), P(m + 3), P(m + 5),
                P(m + 1), P(m + 0), P(m + 2), hbuf, resbuf, Hpost, BT);
        }
        int kd = KDA0 + 11 * l;
        gemm(hbuf, P(kd + 3), nullptr, qlin,  BT, 32, 16, 0, stream);  // Wq
        gemm(hbuf, P(kd + 1), nullptr, klin,  BT, 32, 16, 0, stream);  // Wk
        gemm(hbuf, P(kd + 5), nullptr, vbuf,  BT, 32, 16, 1, stream);  // silu(Wv)
        gemm(hbuf, P(kd + 7), nullptr, AG,    BT, 32, 25, 1, stream);  // silu(a_gate)
        gemm(hbuf, P(kd + 8), nullptr, AU,    BT, 32, 25, 0, stream);  // a_up
        mul2_kernel<<<nblk(BT * 25), 256, 0, stream>>>(AG, AU, AG, BT * 25);
        gemm(AG,   P(kd + 6), nullptr, alphab, BT, 25, 32, 2, stream); // sigmoid(a_down)
        gemm(hbuf, P(kd + 0), nullptr, DG,    BT, 32,  9, 1, stream);  // silu(Wd)
        gemm(DG,   P(kd + 4), nullptr, G2,    BT,  9, 16, 2, stream);  // sigmoid(Wu)
        pope_kernel<<<nblk(BT), 256, 0, stream>>>(qlin, nullptr,   qb, BT, TT);
        pope_kernel<<<nblk(BT), 256, 0, stream>>>(klin, P(kd + 9), kb, BT, TT);
        scan_kernel<<<BB, 512, 0, stream>>>(qb, kb, vbuf, alphab, scano, TT);
        ob_kernel<<<nblk(BT), 256, 0, stream>>>(scano, P(kd + 10), G2, obuf, BT);
        gemm(obuf, P(kd + 2), nullptr, o32, BT, 16, 32, 0, stream);    // Wout
        mix_kernel<<<nblk(BT * 128), 256, 0, stream>>>(resbuf, Hpost, o32, strmbuf, BT);

        // ================= SwiGLU sublayer =================
        rmsnorm_kernel<<<nblk(BT * 4), 256, 0, stream>>>(
            strmbuf, P(SNW0 + 2 * l + 1), snorm, BT * 4, 32, EPS_RMS);
        {
            int m = MHC0 + 9 * (2 * l + 1);
            mhc_kernel<<<nblk(BT), 256, 0, stream>>>(
                snorm, P(m + 7), P(m + 6), P(m + 8), P(m + 4), P(m + 3), P(m + 5),
                P(m + 1), P(m + 0), P(m + 2), hbuf, resbuf, Hpost, BT);
        }
        int sw = SWI0 + 6 * l;
        rmsnorm_kernel<<<nblk(BT), 256, 0, stream>>>(hbuf, P(sw + 2), hn, BT, 32, EPS_RMS);
        gemm(hn, P(sw + 4), nullptr, gd,   BT, 32, 32, 1, stream);     // silu(wd)
        gemm(gd, P(sw + 5), nullptr, gbuf, BT, 32, 51, 2, stream);     // sigmoid(wu)
        gemm(hn, P(sw + 1), nullptr, t1,   BT, 32, 51, 1, stream);     // silu(gate)
        gemm(hn, P(sw + 3), nullptr, t2,   BT, 32, 51, 0, stream);     // up
        mul3_kernel<<<nblk(BT * 51), 256, 0, stream>>>(gbuf, t1, t2, t1, BT * 51);
        gemm(t1, P(sw + 0), nullptr, o32, BT, 51, 32, 0, stream);      // down
        mix_kernel<<<nblk(BT * 128), 256, 0, stream>>>(resbuf, Hpost, o32, strmbuf, BT);

        // ================= AttnRes depth aggregation =================
        for (int a2 = 0; a2 <= l; ++a2)
            gemm(accb + (size_t)a2 * BT * 128, P(PREH0 + l), nullptr,
                 Vbuf + (size_t)a2 * BT * 32, BT, 128, 32, 0, stream);
        attnres_kernel<<<nblk(BT), 256, 0, stream>>>(
            Vbuf, l + 1, P(WW0 + l), P(1), strmbuf, 0, BT);
        hipMemcpyAsync(accb + (size_t)(l + 1) * BT * 128, strmbuf,
                       (size_t)BT * 128 * sizeof(float),
                       hipMemcpyDeviceToDevice, stream);
    }

    // ---- final attn_res over all 9 acc snapshots (layer-7 params) ----
    for (int a2 = 0; a2 < 9; ++a2)
        gemm(accb + (size_t)a2 * BT * 128, P(PREH0 + 7), nullptr,
             Vbuf + (size_t)a2 * BT * 32, BT, 128, 32, 0, stream);
    attnres_kernel<<<nblk(BT), 256, 0, stream>>>(
        Vbuf, 9, P(WW0 + 7), P(1), hout, 1, BT);

    // ---- head ----
    gemm(hout, P(4), nullptr, gbuf, BT, 32, 51, 1, stream);            // silu(head_gate)
    gemm(hout, P(5), nullptr, t2,   BT, 32, 51, 0, stream);            // head_up
    mul2_kernel<<<nblk(BT * 51), 256, 0, stream>>>(gbuf, t2, gbuf, BT * 51);
    gemm(gbuf, P(3), nullptr, (float*)d_out, BT, 51, 11, 0, stream);   // head_down
#undef P
}
